// MVGLayer_18253611008866
// MI455X (gfx1250) — compile-verified, matrix operand order checked by reading, not run on hardware
//
#include <hip/hip_runtime.h>

// ---- types ----
typedef __attribute__((ext_vector_type(2)))  unsigned int u32x2;
typedef __attribute__((ext_vector_type(4)))  unsigned int u32x4;
typedef __attribute__((ext_vector_type(8)))  unsigned int u32x8;
typedef __attribute__((ext_vector_type(4)))  float        f32x4;
typedef __attribute__((ext_vector_type(8)))  float        v8f;
typedef __attribute__((ext_vector_type(2)))  __bf16       bf16x2;
typedef __attribute__((ext_vector_type(16))) __bf16       v16bf;

// ---- problem constants ----
#define BT 8192
#define NT 4096
#define MT 4096

// ---- tiling ----
#define TM 128
#define TN 128
#define KT 32
#define KTILES (NT / KT)  // 128
#define RA 20             // padded LDS stride in u32 (80B)

// Split a pair of f32 into packed bf16 hi parts and packed bf16 residuals.
__device__ __forceinline__ void split_pair(float a, float b,
                                           unsigned& hiP, unsigned& loP) {
  const __bf16 ha = (__bf16)a, hb = (__bf16)b;
  bf16x2 hp; hp[0] = ha; hp[1] = hb;
  hiP = __builtin_bit_cast(unsigned, hp);
  const float ra = a - (float)ha;
  const float rb = b - (float)hb;
  bf16x2 lp; lp[0] = (__bf16)ra; lp[1] = (__bf16)rb;
  loP = __builtin_bit_cast(unsigned, lp);
}

// Raw LDS byte offset of a __shared__ object (generic -> addrspace(3) -> int).
__device__ __forceinline__ unsigned lds_off(const void* p) {
  return (unsigned)(uintptr_t)(__attribute__((address_space(3))) const void*)p;
}

// One 16B async global->LDS copy per lane (ASYNCcnt-tracked).
__device__ __forceinline__ void async_cp16(unsigned ldsByte, unsigned gByte,
                                           unsigned long long base) {
  asm volatile("global_load_async_to_lds_b128 %0, %1, %2"
               :: "v"(ldsByte), "v"(gByte), "s"(base) : "memory");
}
__device__ __forceinline__ void wait_async0() {
  asm volatile("s_wait_asynccnt 0x0" ::: "memory");
}

// =====================================================================
// Pre-pass 1: split x into packed bf16 k-pair chunks
// layout: [rowBlock][kt][row 0..127][kp 0..15] u32, 8KB per (rowBlock,kt)
// =====================================================================
__global__ __launch_bounds__(256) void prepass_x(
    const float* __restrict__ x, unsigned* __restrict__ xH, unsigned* __restrict__ xL)
{
  const size_t gid = (size_t)blockIdx.x * 256 + threadIdx.x;  // BT*NT/4 units
  const int j   = (int)(gid & 1023);   // f32x4 index within row
  const int row = (int)(gid >> 10);
  const f32x4 xa = *reinterpret_cast<const f32x4*>(x + ((size_t)row << 12) + 4 * j);
  unsigned h0, l0, h1, l1;
  split_pair(xa.x, xa.y, h0, l0);
  split_pair(xa.z, xa.w, h1, l1);
  const int kt = j >> 3, jj = j & 7;
  const size_t dst =
      ((size_t)(((row >> 7) * KTILES + kt) * 128 + (row & 127)) << 4) + 2 * jj;
  u32x2 ph; ph[0] = h0; ph[1] = h1;
  u32x2 pl; pl[0] = l0; pl[1] = l1;
  *reinterpret_cast<u32x2*>(xH + dst) = ph;
  *reinterpret_cast<u32x2*>(xL + dst) = pl;
}

// =====================================================================
// Pre-pass 2: fuse W = Wm + e^{.5u} * eps * e^{.5v}, split, store
// COLUMN-major packed k-pairs: [colBlock][kt][col 0..127][kp 0..15] u32
// =====================================================================
__global__ __launch_bounds__(256) void prepass_w(
    const float* __restrict__ Wm, const float* __restrict__ Wu,
    const float* __restrict__ Wv, const float* __restrict__ eps,
    unsigned* __restrict__ wH, unsigned* __restrict__ wL)
{
  const size_t gid = (size_t)blockIdx.x * 256 + threadIdx.x;  // MT*KTILES units
  const int m  = (int)(gid & (MT - 1));
  const int kt = (int)(gid >> 12);
  const float sv = __expf(0.5f * Wv[m]);
  unsigned H[16], L[16];
#pragma unroll
  for (int kk = 0; kk < 16; ++kk) {
    const int k = kt * KT + 2 * kk;
    const float su0 = __expf(0.5f * Wu[k]);
    const float su1 = __expf(0.5f * Wu[k + 1]);
    const float w0 = Wm[(size_t)k * MT + m]       + su0 * eps[(size_t)k * MT + m]       * sv;
    const float w1 = Wm[(size_t)(k + 1) * MT + m] + su1 * eps[(size_t)(k + 1) * MT + m] * sv;
    split_pair(w0, w1, H[kk], L[kk]);
  }
  const size_t dst = ((size_t)(((m >> 7) * KTILES + kt) * 128 + (m & 127)) << 4);
#pragma unroll
  for (int q = 0; q < 4; ++q) {
    u32x4 hv, lv;
#pragma unroll
    for (int r = 0; r < 4; ++r) { hv[r] = H[4 * q + r]; lv[r] = L[4 * q + r]; }
    *reinterpret_cast<u32x4*>(wH + dst + 4 * q) = hv;
    *reinterpret_cast<u32x4*>(wL + dst + 4 * q) = lv;
  }
}

// =====================================================================
// Main GEMM on pre-packed bf16 hi/lo: async global->LDS staging,
// zero conversion VALU in the hot loop, 3 WMMAs per logical fp32 tile.
// =====================================================================
__global__ __launch_bounds__(256) void gemm_packed(
    const unsigned* __restrict__ xH, const unsigned* __restrict__ xL,
    const unsigned* __restrict__ wH, const unsigned* __restrict__ wL,
    float* __restrict__ out)
{
  __shared__ unsigned AsHi[TM * RA];
  __shared__ unsigned AsLo[TM * RA];
  __shared__ unsigned WsHi[TN * RA];
  __shared__ unsigned WsLo[TN * RA];

  const int tid   = threadIdx.x;
  const int lane  = tid & 31;
  const int lo    = lane & 15;
  const int hi    = lane >> 4;
  const int hi4   = hi * 4;
  const int wave  = tid >> 5;
  const int wmRow = (wave >> 1) * 32;
  const int wnCol = (wave & 1) * 64;
  const int blockRow = blockIdx.y * TM;
  const int blockCol = blockIdx.x * TN;

  // ---- async copy geometry: 512 16B units per 8KB chunk, 2 per thread ----
  unsigned lAH[2], lAL[2], lWH[2], lWL[2], gA[2], gW[2];
#pragma unroll
  for (int t = 0; t < 2; ++t) {
    const int u   = tid + 256 * t;
    const unsigned pad = (unsigned)((u >> 2) * 80 + (u & 3) * 16);  // padded LDS
    lAH[t] = lds_off(AsHi) + pad;
    lAL[t] = lds_off(AsLo) + pad;
    lWH[t] = lds_off(WsHi) + pad;
    lWL[t] = lds_off(WsLo) + pad;
    gA[t]  = (unsigned)((size_t)blockIdx.y * KTILES * 8192) + (unsigned)u * 16u;
    gW[t]  = (unsigned)((size_t)blockIdx.x * KTILES * 8192) + (unsigned)u * 16u;
  }
  const unsigned long long bXH = (unsigned long long)xH;
  const unsigned long long bXL = (unsigned long long)xL;
  const unsigned long long bWH = (unsigned long long)wH;
  const unsigned long long bWL = (unsigned long long)wL;

  // ---- fragment base pointers ----
  const unsigned* aBaseHi[2]; const unsigned* aBaseLo[2];
#pragma unroll
  for (int mi = 0; mi < 2; ++mi) {
    const int row = wmRow + mi * 16 + lo;
    aBaseHi[mi] = &AsHi[row * RA];
    aBaseLo[mi] = &AsLo[row * RA];
  }
  const unsigned* bBaseHi[4]; const unsigned* bBaseLo[4];
#pragma unroll
  for (int ni = 0; ni < 4; ++ni) {
    const int cb = wnCol + ni * 16 + lo;
    bBaseHi[ni] = &WsHi[cb * RA];
    bBaseLo[ni] = &WsLo[cb * RA];
  }

  v8f acc[2][4] = {};

  // ---- prologue: async-stage tile 0 ----
#pragma unroll
  for (int t = 0; t < 2; ++t) {
    async_cp16(lAH[t], gA[t], bXH);
    async_cp16(lAL[t], gA[t], bXL);
    async_cp16(lWH[t], gW[t], bWH);
    async_cp16(lWL[t], gW[t], bWL);
  }

#pragma unroll 1
  for (int kt = 0; kt < KTILES; ++kt) {
    wait_async0();      // own async copies done
    __syncthreads();    // => whole tile visible

    // ---- hi fragments first; start hi*hi WMMAs as data arrives ----
    u32x8 aHiF[2], bHiF[4];
#pragma unroll
    for (int mi = 0; mi < 2; ++mi) {
      const u32x4 h0 = *reinterpret_cast<const u32x4*>(aBaseHi[mi] + hi4);
      const u32x4 h1 = *reinterpret_cast<const u32x4*>(aBaseHi[mi] + 8 + hi4);
      aHiF[mi] = __builtin_shufflevector(h0, h1, 0, 1, 2, 3, 4, 5, 6, 7);
    }
#pragma unroll
    for (int ni = 0; ni < 4; ++ni) {
      const u32x4 b0 = *reinterpret_cast<const u32x4*>(bBaseHi[ni] + hi4);
      const u32x4 b1 = *reinterpret_cast<const u32x4*>(bBaseHi[ni] + 8 + hi4);
      bHiF[ni] = __builtin_shufflevector(b0, b1, 0, 1, 2, 3, 4, 5, 6, 7);
    }
#pragma unroll
    for (int ni = 0; ni < 4; ++ni) {
      const v16bf bh = __builtin_bit_cast(v16bf, bHiF[ni]);
#pragma unroll
      for (int mi = 0; mi < 2; ++mi) {
        const v16bf ah = __builtin_bit_cast(v16bf, aHiF[mi]);
        acc[mi][ni] = __builtin_amdgcn_wmma_f32_16x16x32_bf16(
            false, ah, false, bh, (short)0, acc[mi][ni], false, false);
      }
    }

    // ---- lo fragments ----
    u32x8 aLoF[2], bLoF[4];
#pragma unroll
    for (int mi = 0; mi < 2; ++mi) {
      const u32x4 q0 = *reinterpret_cast<const u32x4*>(aBaseLo[mi] + hi4);
      const u32x4 q1 = *reinterpret_cast<const u32x4*>(aBaseLo[mi] + 8 + hi4);
      aLoF[mi] = __builtin_shufflevector(q0, q1, 0, 1, 2, 3, 4, 5, 6, 7);
    }
#pragma unroll
    for (int ni = 0; ni < 4; ++ni) {
      const u32x4 c0 = *reinterpret_cast<const u32x4*>(bBaseLo[ni] + hi4);
      const u32x4 c1 = *reinterpret_cast<const u32x4*>(bBaseLo[ni] + 8 + hi4);
      bLoF[ni] = __builtin_shufflevector(c0, c1, 0, 1, 2, 3, 4, 5, 6, 7);
    }

    __syncthreads();    // all waves done reading LDS (dscnt fenced)

    // ---- issue next tile's async copies; they overlap the WMMAs below ----
    if (kt + 1 < KTILES) {
#pragma unroll
      for (int t = 0; t < 2; ++t) {
        gA[t] += 8192u; gW[t] += 8192u;
        async_cp16(lAH[t], gA[t], bXH);
        async_cp16(lAL[t], gA[t], bXL);
        async_cp16(lWH[t], gW[t], bWH);
        async_cp16(lWL[t], gW[t], bWL);
      }
    }

    // ---- remaining 16 WMMAs: hi*lo + lo*hi ----
#pragma unroll
    for (int ni = 0; ni < 4; ++ni) {
      const v16bf bh = __builtin_bit_cast(v16bf, bHiF[ni]);
      const v16bf bl = __builtin_bit_cast(v16bf, bLoF[ni]);
#pragma unroll
      for (int mi = 0; mi < 2; ++mi) {
        const v16bf ah = __builtin_bit_cast(v16bf, aHiF[mi]);
        const v16bf al = __builtin_bit_cast(v16bf, aLoF[mi]);
        acc[mi][ni] = __builtin_amdgcn_wmma_f32_16x16x32_bf16(
            false, ah, false, bl, (short)0, acc[mi][ni], false, false);
        acc[mi][ni] = __builtin_amdgcn_wmma_f32_16x16x32_bf16(
            false, al, false, bh, (short)0, acc[mi][ni], false, false);
      }
    }
  }

#pragma unroll
  for (int mi = 0; mi < 2; ++mi) {
#pragma unroll
    for (int ni = 0; ni < 4; ++ni) {
      const int col   = blockCol + wnCol + ni * 16 + lo;
      const int rbase = blockRow + wmRow + mi * 16 + hi * 8;
#pragma unroll
      for (int r = 0; r < 8; ++r)
        out[(size_t)(rbase + r) * MT + col] = acc[mi][ni][r];
    }
  }
}

// =====================================================================
// Fallback: fused single-kernel path (used when d_ws is too small)
// =====================================================================
__global__ __launch_bounds__(256) void mvg_fused_gemm(
    const float* __restrict__ x, const float* __restrict__ Wm,
    const float* __restrict__ Wu, const float* __restrict__ Wv,
    const float* __restrict__ eps, float* __restrict__ out)
{
  __shared__ unsigned AsHi[TM * RA];
  __shared__ unsigned AsLo[TM * RA];
  __shared__ unsigned WsHi[TN * RA];
  __shared__ unsigned WsLo[TN * RA];
  __shared__ float su_s[NT];
  __shared__ float sv_s[TN];

  const int tid   = threadIdx.x;
  const int lane  = tid & 31;
  const int lo    = lane & 15;
  const int hi    = lane >> 4;
  const int hi4   = hi * 4;
  const int wave  = tid >> 5;
  const int wmRow = (wave >> 1) * 32;
  const int wnCol = (wave & 1) * 64;
  const int blockRow = blockIdx.y * TM;
  const int blockCol = blockIdx.x * TN;

  for (int i = tid; i < NT; i += 256) su_s[i] = __expf(0.5f * Wu[i]);
  if (tid < TN) sv_s[tid] = __expf(0.5f * Wv[blockCol + tid]);

  int aRow[4], aJ[4]; size_t aOff[4];
#pragma unroll
  for (int t = 0; t < 4; ++t) {
    const int idx = tid + 256 * t;
    aRow[t] = idx >> 3; aJ[t] = idx & 7;
    aOff[t] = (size_t)(blockRow + aRow[t]) * NT + 4 * aJ[t];
  }
  int wKp[2], wN4[2]; size_t wOff[2];
#pragma unroll
  for (int t = 0; t < 2; ++t) {
    const int u = tid + 256 * t;
    wKp[t] = u >> 5; wN4[t] = (u & 31) * 4;
    wOff[t] = (size_t)(2 * wKp[t]) * MT + blockCol + wN4[t];
  }
  const unsigned* aBaseHi[2]; const unsigned* aBaseLo[2];
#pragma unroll
  for (int mi = 0; mi < 2; ++mi) {
    const int row = wmRow + mi * 16 + lo;
    aBaseHi[mi] = &AsHi[row * RA]; aBaseLo[mi] = &AsLo[row * RA];
  }
  const unsigned* bBaseHi[4]; const unsigned* bBaseLo[4];
#pragma unroll
  for (int ni = 0; ni < 4; ++ni) {
    const int cb = wnCol + ni * 16 + lo;
    bBaseHi[ni] = &WsHi[cb * RA]; bBaseLo[ni] = &WsLo[cb * RA];
  }

  f32x4 pA[4], pWm0[2], pWm1[2], pE0[2], pE1[2];
#pragma unroll
  for (int t = 0; t < 4; ++t) pA[t] = *reinterpret_cast<const f32x4*>(x + aOff[t]);
#pragma unroll
  for (int t = 0; t < 2; ++t) {
    pWm0[t] = *reinterpret_cast<const f32x4*>(Wm  + wOff[t]);
    pWm1[t] = *reinterpret_cast<const f32x4*>(Wm  + wOff[t] + MT);
    pE0[t]  = *reinterpret_cast<const f32x4*>(eps + wOff[t]);
    pE1[t]  = *reinterpret_cast<const f32x4*>(eps + wOff[t] + MT);
  }

  v8f acc[2][4] = {};
  __syncthreads();

#pragma unroll 1
  for (int kt = 0; kt < KTILES; ++kt) {
    const int k0 = kt * KT;
#pragma unroll
    for (int t = 0; t < 4; ++t) {
      const f32x4 xa = pA[t];
      unsigned ph0, pl0, ph1, pl1;
      split_pair(xa.x, xa.y, ph0, pl0);
      split_pair(xa.z, xa.w, ph1, pl1);
      u32x2 ph; ph[0] = ph0; ph[1] = ph1;
      u32x2 pl; pl[0] = pl0; pl[1] = pl1;
      *reinterpret_cast<u32x2*>(&AsHi[aRow[t] * RA + 2 * aJ[t]]) = ph;
      *reinterpret_cast<u32x2*>(&AsLo[aRow[t] * RA + 2 * aJ[t]]) = pl;
    }
#pragma unroll
    for (int t = 0; t < 2; ++t) {
      const int kg = k0 + 2 * wKp[t];
      const f32x4 sv = *reinterpret_cast<const f32x4*>(&sv_s[wN4[t]]);
      const float su0 = su_s[kg], su1 = su_s[kg + 1];
      const f32x4 w0 = pWm0[t] + su0 * pE0[t] * sv;
      const f32x4 w1 = pWm1[t] + su1 * pE1[t] * sv;
#pragma unroll
      for (int i = 0; i < 4; ++i) {
        unsigned ph, pl;
        split_pair(w0[i], w1[i], ph, pl);
        WsHi[(wN4[t] + i) * RA + wKp[t]] = ph;
        WsLo[(wN4[t] + i) * RA + wKp[t]] = pl;
      }
    }
    {
      const int ktn = (kt + 1 < KTILES) ? (kt + 1) : (KTILES - 1);
      const size_t dA = (size_t)ktn * KT;
      const size_t dW = (size_t)ktn * KT * MT;
#pragma unroll
      for (int t = 0; t < 4; ++t)
        pA[t] = *reinterpret_cast<const f32x4*>(x + aOff[t] + dA);
#pragma unroll
      for (int t = 0; t < 2; ++t) {
        pWm0[t] = *reinterpret_cast<const f32x4*>(Wm  + wOff[t] + dW);
        pWm1[t] = *reinterpret_cast<const f32x4*>(Wm  + wOff[t] + dW + MT);
        pE0[t]  = *reinterpret_cast<const f32x4*>(eps + wOff[t] + dW);
        pE1[t]  = *reinterpret_cast<const f32x4*>(eps + wOff[t] + dW + MT);
      }
    }
    __syncthreads();
    u32x8 aHi[2], aLo[2];
#pragma unroll
    for (int mi = 0; mi < 2; ++mi) {
      const u32x4 h0 = *reinterpret_cast<const u32x4*>(aBaseHi[mi] + hi4);
      const u32x4 h1 = *reinterpret_cast<const u32x4*>(aBaseHi[mi] + 8 + hi4);
      const u32x4 q0 = *reinterpret_cast<const u32x4*>(aBaseLo[mi] + hi4);
      const u32x4 q1 = *reinterpret_cast<const u32x4*>(aBaseLo[mi] + 8 + hi4);
      aHi[mi] = __builtin_shufflevector(h0, h1, 0, 1, 2, 3, 4, 5, 6, 7);
      aLo[mi] = __builtin_shufflevector(q0, q1, 0, 1, 2, 3, 4, 5, 6, 7);
    }
#pragma unroll
    for (int ni = 0; ni < 4; ++ni) {
      const u32x4 b0 = *reinterpret_cast<const u32x4*>(bBaseHi[ni] + hi4);
      const u32x4 b1 = *reinterpret_cast<const u32x4*>(bBaseHi[ni] + 8 + hi4);
      const u32x4 c0 = *reinterpret_cast<const u32x4*>(bBaseLo[ni] + hi4);
      const u32x4 c1 = *reinterpret_cast<const u32x4*>(bBaseLo[ni] + 8 + hi4);
      const v16bf bh = __builtin_bit_cast(v16bf, __builtin_shufflevector(b0, b1, 0, 1, 2, 3, 4, 5, 6, 7));
      const v16bf bl = __builtin_bit_cast(v16bf, __builtin_shufflevector(c0, c1, 0, 1, 2, 3, 4, 5, 6, 7));
#pragma unroll
      for (int mi = 0; mi < 2; ++mi) {
        const v16bf ah = __builtin_bit_cast(v16bf, aHi[mi]);
        const v16bf al = __builtin_bit_cast(v16bf, aLo[mi]);
        acc[mi][ni] = __builtin_amdgcn_wmma_f32_16x16x32_bf16(
            false, ah, false, bh, (short)0, acc[mi][ni], false, false);
        acc[mi][ni] = __builtin_amdgcn_wmma_f32_16x16x32_bf16(
            false, ah, false, bl, (short)0, acc[mi][ni], false, false);
        acc[mi][ni] = __builtin_amdgcn_wmma_f32_16x16x32_bf16(
            false, al, false, bh, (short)0, acc[mi][ni], false, false);
      }
    }
    __syncthreads();
  }

#pragma unroll
  for (int mi = 0; mi < 2; ++mi) {
#pragma unroll
    for (int ni = 0; ni < 4; ++ni) {
      const int col   = blockCol + wnCol + ni * 16 + lo;
      const int rbase = blockRow + wmRow + mi * 16 + hi * 8;
#pragma unroll
      for (int r = 0; r < 8; ++r)
        out[(size_t)(rbase + r) * MT + col] = acc[mi][ni][r];
    }
  }
}

extern "C" void kernel_launch(void* const* d_in, const int* in_sizes, int n_in,
                              void* d_out, int out_size, void* d_ws, size_t ws_size,
                              hipStream_t stream) {
  (void)in_sizes; (void)n_in; (void)out_size;
  const float* x  = (const float*)d_in[0];
  const float* Wm = (const float*)d_in[1];
  const float* Wu = (const float*)d_in[2];
  const float* Wv = (const float*)d_in[3];
  const float* ep = (const float*)d_in[4];

  const size_t xPackU32 = (size_t)BT * NT / 2;   // 16M u32 = 64MB per array
  const size_t wPackU32 = (size_t)NT * MT / 2;   // 8M u32 = 32MB per array
  const size_t wsNeeded = (2 * xPackU32 + 2 * wPackU32) * sizeof(unsigned);  // 192MB

  if (ws_size >= wsNeeded) {
    unsigned* xH = (unsigned*)d_ws;
    unsigned* xL = xH + xPackU32;
    unsigned* wH = xL + xPackU32;
    unsigned* wL = wH + wPackU32;
    prepass_x<<<dim3((unsigned)((size_t)BT * NT / 4 / 256)), dim3(256), 0, stream>>>(x, xH, xL);
    prepass_w<<<dim3((unsigned)((size_t)MT * KTILES / 256)), dim3(256), 0, stream>>>(Wm, Wu, Wv, ep, wH, wL);
    gemm_packed<<<dim3(MT / TN, BT / TM), dim3(256), 0, stream>>>(xH, xL, wH, wL, (float*)d_out);
  } else {
    mvg_fused_gemm<<<dim3(MT / TN, BT / TM), dim3(256), 0, stream>>>(x, Wm, Wu, Wv, ep, (float*)d_out);
  }
}
